// EnhancedTransformerEncoder_11854109737598
// MI455X (gfx1250) — matrix operand order repaired and checked by reading, not compiled, not run on hardware
//
#include <hip/hip_runtime.h>
#include <hip/hip_bf16.h>

// ---------------------------------------------------------------------------
// EnhancedTransformerEncoder forward for MI455X (gfx1250, wave32, WMMA).
// All GEMMs use v_wmma_f32_16x16x32_f16. Weights are converted once to f16
// and stored TRANSPOSED (Wt[n][k]) so every B-fragment is two contiguous
// 128-bit loads per lane. V is written per-head transposed so attn@V also
// gets contiguous B loads. Energy (QK^T) has K=hd=32 -> exactly one WMMA
// per 16x16 tile, with K's natural [s,hd] layout already matching the
// B-fragment layout (no transpose needed at all).
// ---------------------------------------------------------------------------

typedef __attribute__((ext_vector_type(16))) _Float16 v16h;
typedef __attribute__((ext_vector_type(8)))  _Float16 v8h;
typedef __attribute__((ext_vector_type(8)))  float    v8f;

#define Bc   4
#define Sc   2048
#define DM   256
#define Hh   8
#define HD   32
#define DFF  512
#define EPSL 1e-5f

// ---------------------------------------------------------------------------
// Weight convert + transpose: Wt[n*Kd + k] = (f16) W[k*N + n]
// ---------------------------------------------------------------------------
__global__ void convert_t_kernel(const float* __restrict__ W,
                                 _Float16* __restrict__ Wt,
                                 int Kd, int N) {
    int idx = blockIdx.x * 256 + threadIdx.x;
    int total = Kd * N;
    if (idx < total) {
        int n = idx / Kd;
        int k = idx - n * Kd;
        Wt[idx] = (_Float16)W[k * N + n];
    }
}

// ---------------------------------------------------------------------------
// Embedding: emb = src @ We + be  (K=16, scalar FMA; trivial FLOPs)
// Writes f32 (for residual) and f16 (for WMMA) copies.
// ---------------------------------------------------------------------------
__global__ void embed_kernel(const float* __restrict__ src,
                             const float* __restrict__ We,
                             const float* __restrict__ be,
                             float* __restrict__ embF,
                             _Float16* __restrict__ embH) {
    int idx = blockIdx.x * 256 + threadIdx.x;   // 0 .. B*S*DM-1
    int m = idx >> 8;                           // row (b*S+s)
    int n = idx & (DM - 1);                     // col
    const float* s = src + (size_t)m * 16;
    float acc = be[n];
#pragma unroll
    for (int k = 0; k < 16; ++k) acc = fmaf(s[k], We[k * DM + n], acc);
    embF[idx] = acc;
    embH[idx] = (_Float16)acc;
}

// ---------------------------------------------------------------------------
// Generic WMMA GEMM: C[M,N] = A[M,Kd] (f16) @ Bt[N,Kd]^T (f16) + bias.
// 8 waves/block, one 16x16 tile per wave, exact grids -> EXEC all ones.
// Epilogue modes:
//   0: f16 out, per-head [b,h,s,d]   (Q,K projections)
//   1: f16 out, per-head transposed [b,h,d,s]  (V projection)
//   3: f16 out, row-major M x N, ReLU          (FFN layer 1)
//   4: f32 out, row-major M x N                (O-proj, FFN layer 2)
// ---------------------------------------------------------------------------
__global__ void gemm16_kernel(const _Float16* __restrict__ A,
                              const _Float16* __restrict__ Bt,
                              const float* __restrict__ bias,
                              _Float16* __restrict__ outH,
                              float* __restrict__ outF,
                              int M, int N, int Kd, int mode) {
    int lane = threadIdx.x & 31;
    int wave = threadIdx.x >> 5;
    int ntiles = N >> 4;
    int tile = blockIdx.x * 8 + wave;
    int mt = tile / ntiles;
    int nt = tile - mt * ntiles;
    int mbase = mt << 4, nbase = nt << 4;

    // A fragment (16-bit A 16x32, ISA layout)
    int arow = lane & 15;
    int akb  = (lane >> 4) << 3;   // 0 or 8
    // B fragment (16-bit B 32x16, ISA layout)
    int bcol = lane & 15;
    int bkb  = (lane >> 4) << 3;   // 0 or 8

    const _Float16* ap = A  + (size_t)(mbase + arow) * Kd + akb;
    const _Float16* bp = Bt + (size_t)(nbase + bcol) * Kd + bkb;

    v8f acc = {};
    for (int k0 = 0; k0 < Kd; k0 += 32) {
        v8h alo = *(const v8h*)(ap + k0);        // K = akb..akb+7
        v8h ahi = *(const v8h*)(ap + k0 + 16);   // K = 16+akb..16+akb+7
        v8h blo = *(const v8h*)(bp + k0);        // K = bkb..bkb+7
        v8h bhi = *(const v8h*)(bp + k0 + 16);   // K = 16+bkb..16+bkb+7
        v16h a, b;
#pragma unroll
        for (int i = 0; i < 8; ++i) {
            a[i] = alo[i]; a[i + 8] = ahi[i];
            b[i] = blo[i]; b[i + 8] = bhi[i];
        }
        acc = __builtin_amdgcn_wmma_f32_16x16x32_f16(
            false, a, false, b, (short)0, acc, false, false);
    }

    // C/D layout: VGPR i -> M = i (lanes 0-15) / i+8 (lanes 16-31), N = lane&15
    int col   = lane & 15;
    int rbase = (lane >> 4) << 3;
    int n = nbase + col;
    float bsv = bias ? bias[n] : 0.0f;

    if (mode == 0) {              // per-head Q/K layout [b,h,s,hd]
        int h = n >> 5, d = n & 31;
#pragma unroll
        for (int i = 0; i < 8; ++i) {
            int m = mbase + rbase + i;
            int b = m >> 11, s = m & (Sc - 1);
            outH[(((size_t)b * Hh + h) * Sc + s) * HD + d] = (_Float16)(acc[i] + bsv);
        }
    } else if (mode == 1) {       // per-head transposed V layout [b,h,hd,s]
        int h = n >> 5, d = n & 31;
#pragma unroll
        for (int i = 0; i < 8; ++i) {
            int m = mbase + rbase + i;
            int b = m >> 11, s = m & (Sc - 1);
            outH[(((size_t)b * Hh + h) * HD + d) * Sc + s] = (_Float16)(acc[i] + bsv);
        }
    } else if (mode == 3) {       // ReLU f16
#pragma unroll
        for (int i = 0; i < 8; ++i) {
            int m = mbase + rbase + i;
            float v = acc[i] + bsv;
            outH[(size_t)m * N + n] = (_Float16)(v > 0.0f ? v : 0.0f);
        }
    } else {                      // mode 4: f32 row-major
#pragma unroll
        for (int i = 0; i < 8; ++i) {
            int m = mbase + rbase + i;
            outF[(size_t)m * N + n] = acc[i] + bsv;
        }
    }
}

// ---------------------------------------------------------------------------
// Energy: attn[b,h,q,k] = (Q . K) / sqrt(hd).  K-dim = 32 -> ONE wmma/tile.
// K's natural [s,hd] row-major layout IS the B fragment layout (Kt rows).
// grid = (S/16, S/16/8, B*H), 8 waves/block each own 16x16 k-tile.
// ---------------------------------------------------------------------------
__global__ void energy_kernel(const _Float16* __restrict__ Q,
                              const _Float16* __restrict__ Km,
                              float* __restrict__ attn) {
    int lane = threadIdx.x & 31;
    int wave = threadIdx.x >> 5;
    int bh = blockIdx.z;
    int qbase = blockIdx.x << 4;
    int kbase = ((blockIdx.y << 3) + wave) << 4;

    const _Float16* Qh = Q  + (size_t)bh * Sc * HD;
    const _Float16* Kh = Km + (size_t)bh * Sc * HD;

    int arow = lane & 15, akb = (lane >> 4) << 3;
    int bcol = lane & 15, bkb = (lane >> 4) << 3;
    const _Float16* ap = Qh + (size_t)(qbase + arow) * HD + akb;
    const _Float16* bp = Kh + (size_t)(kbase + bcol) * HD + bkb;

    v8h alo = *(const v8h*)(ap);
    v8h ahi = *(const v8h*)(ap + 16);
    v8h blo = *(const v8h*)(bp);
    v8h bhi = *(const v8h*)(bp + 16);
    v16h a, b;
#pragma unroll
    for (int i = 0; i < 8; ++i) {
        a[i] = alo[i]; a[i + 8] = ahi[i];
        b[i] = blo[i]; b[i + 8] = bhi[i];
    }
    v8f acc = {};
    acc = __builtin_amdgcn_wmma_f32_16x16x32_f16(
        false, a, false, b, (short)0, acc, false, false);

    const float scale = 0.17677669529663687f;   // 1/sqrt(32)
    int col = lane & 15;
    int rbase = (lane >> 4) << 3;
    float* dst = attn + ((size_t)bh * Sc + qbase) * Sc + kbase;
#pragma unroll
    for (int i = 0; i < 8; ++i)
        dst[(size_t)(rbase + i) * Sc + col] = acc[i] * scale;
}

// ---------------------------------------------------------------------------
// Row softmax over 2048 elements, in place. One block (8 waves) per row.
// ---------------------------------------------------------------------------
__global__ void softmax_kernel(float* __restrict__ attn) {
    __shared__ float red[256];
    float* p = attn + (size_t)blockIdx.x * Sc;
    int tid = threadIdx.x;

    float v[8];
#pragma unroll
    for (int j = 0; j < 8; ++j) v[j] = p[tid + j * 256];

    float lmax = v[0];
#pragma unroll
    for (int j = 1; j < 8; ++j) lmax = fmaxf(lmax, v[j]);
    red[tid] = lmax;
    __syncthreads();
    for (int s = 128; s > 0; s >>= 1) {
        if (tid < s) red[tid] = fmaxf(red[tid], red[tid + s]);
        __syncthreads();
    }
    float rmax = red[0];
    __syncthreads();

    float lsum = 0.0f;
#pragma unroll
    for (int j = 0; j < 8; ++j) { v[j] = __expf(v[j] - rmax); lsum += v[j]; }
    red[tid] = lsum;
    __syncthreads();
    for (int s = 128; s > 0; s >>= 1) {
        if (tid < s) red[tid] += red[tid + s];
        __syncthreads();
    }
    float inv = 1.0f / red[0];
#pragma unroll
    for (int j = 0; j < 8; ++j) p[tid + j * 256] = v[j] * inv;
}

// ---------------------------------------------------------------------------
// ctx = attn @ V.  M=S, N=hd=32, Kd=S per (b,h).  A built on the fly from
// f32 attn rows (b128 loads + cvt to f16); B from transposed Vt rows.
// Output written straight into merged-head f16 layout [b,s,h*hd+d].
// ---------------------------------------------------------------------------
__global__ void ctx_kernel(const float* __restrict__ attn,
                           const _Float16* __restrict__ Vt,
                           _Float16* __restrict__ ctxH) {
    int lane = threadIdx.x & 31;
    int wave = threadIdx.x >> 5;
    int bh = blockIdx.z;
    int b = bh >> 3, h = bh & 7;
    int t = blockIdx.x * 8 + wave;      // 0..255 tiles per (b,h)
    int mt = t >> 1, nt = t & 1;
    int mbase = mt << 4, nbase = nt << 4;

    int akb = (lane >> 4) << 3;
    int bkb = (lane >> 4) << 3;
    const float*    Ar = attn + ((size_t)bh * Sc + mbase + (lane & 15)) * Sc;
    const _Float16* Bp = Vt + ((size_t)bh * HD + nbase + (lane & 15)) * Sc + bkb;

    v8f acc = {};
    for (int k0 = 0; k0 < Sc; k0 += 32) {
        float4 f0 = *(const float4*)(Ar + k0 + akb);
        float4 f1 = *(const float4*)(Ar + k0 + akb + 4);
        float4 f2 = *(const float4*)(Ar + k0 + 16 + akb);
        float4 f3 = *(const float4*)(Ar + k0 + 16 + akb + 4);
        v8h blo = *(const v8h*)(Bp + k0);
        v8h bhi = *(const v8h*)(Bp + k0 + 16);
        v16h a, bmat;
        a[0]  = (_Float16)f0.x; a[1]  = (_Float16)f0.y;
        a[2]  = (_Float16)f0.z; a[3]  = (_Float16)f0.w;
        a[4]  = (_Float16)f1.x; a[5]  = (_Float16)f1.y;
        a[6]  = (_Float16)f1.z; a[7]  = (_Float16)f1.w;
        a[8]  = (_Float16)f2.x; a[9]  = (_Float16)f2.y;
        a[10] = (_Float16)f2.z; a[11] = (_Float16)f2.w;
        a[12] = (_Float16)f3.x; a[13] = (_Float16)f3.y;
        a[14] = (_Float16)f3.z; a[15] = (_Float16)f3.w;
#pragma unroll
        for (int i = 0; i < 8; ++i) { bmat[i] = blo[i]; bmat[i + 8] = bhi[i]; }
        acc = __builtin_amdgcn_wmma_f32_16x16x32_f16(
            false, a, false, bmat, (short)0, acc, false, false);
    }

    int col = lane & 15;
    int rbase = (lane >> 4) << 3;
#pragma unroll
    for (int i = 0; i < 8; ++i) {
        int s = mbase + rbase + i;
        ctxH[((size_t)b * Sc + s) * DM + h * HD + nbase + col] = (_Float16)acc[i];
    }
}

// ---------------------------------------------------------------------------
// LayerNorm(xa + xb) * g + b. One block per row of 256. Optional f16 copy.
// ---------------------------------------------------------------------------
__global__ void ln_kernel(const float* __restrict__ xa,
                          const float* __restrict__ xb,
                          const float* __restrict__ g,
                          const float* __restrict__ bt,
                          float* __restrict__ outF,
                          _Float16* __restrict__ outH) {
    __shared__ float red[256];
    size_t base = (size_t)blockIdx.x * DM;
    int tid = threadIdx.x;
    float v = xa[base + tid] + xb[base + tid];

    red[tid] = v;
    __syncthreads();
    for (int s = 128; s > 0; s >>= 1) {
        if (tid < s) red[tid] += red[tid + s];
        __syncthreads();
    }
    float mu = red[0] * (1.0f / DM);
    __syncthreads();

    float d = v - mu;
    red[tid] = d * d;
    __syncthreads();
    for (int s = 128; s > 0; s >>= 1) {
        if (tid < s) red[tid] += red[tid + s];
        __syncthreads();
    }
    float var = red[0] * (1.0f / DM);
    float y = d * rsqrtf(var + EPSL) * g[tid] + bt[tid];
    outF[base + tid] = y;
    if (outH) outH[base + tid] = (_Float16)y;
}

// ---------------------------------------------------------------------------
// Host: orchestrate the pipeline on `stream`. All scratch in d_ws.
// ---------------------------------------------------------------------------
extern "C" void kernel_launch(void* const* d_in, const int* in_sizes, int n_in,
                              void* d_out, int out_size, void* d_ws, size_t ws_size,
                              hipStream_t stream) {
    const float* src = (const float*)d_in[0];
    const float* We  = (const float*)d_in[1];
    const float* be  = (const float*)d_in[2];
    const float* Wq  = (const float*)d_in[3];
    const float* bq  = (const float*)d_in[4];
    const float* Wk  = (const float*)d_in[5];
    const float* bk  = (const float*)d_in[6];
    const float* Wv  = (const float*)d_in[7];
    const float* bv  = (const float*)d_in[8];
    const float* Wo  = (const float*)d_in[9];
    const float* bo  = (const float*)d_in[10];
    const float* g1  = (const float*)d_in[11];
    const float* b1  = (const float*)d_in[12];
    const float* g2  = (const float*)d_in[13];
    const float* b2  = (const float*)d_in[14];
    const float* W1  = (const float*)d_in[15];
    const float* c1  = (const float*)d_in[16];
    const float* W2  = (const float*)d_in[17];
    const float* c2  = (const float*)d_in[18];

    float* outF = (float*)d_out;                         // [B,S,DM] f32
    float* attn = outF + (size_t)Bc * Sc * DM;           // [B,H,S,S] f32

    const size_t MS = (size_t)Bc * Sc;                   // 8192 rows
    char* w = (char*)d_ws;
    auto take = [&](size_t bytes) {
        void* p = (void*)w;
        w += (bytes + 255) & ~(size_t)255;
        return p;
    };
    float*    embF = (float*)   take(MS * DM * 4);
    _Float16* embH = (_Float16*)take(MS * DM * 2);
    _Float16* WqT  = (_Float16*)take((size_t)DM * DM * 2);
    _Float16* WkT  = (_Float16*)take((size_t)DM * DM * 2);
    _Float16* WvT  = (_Float16*)take((size_t)DM * DM * 2);
    _Float16* WoT  = (_Float16*)take((size_t)DM * DM * 2);
    _Float16* W1T  = (_Float16*)take((size_t)DM * DFF * 2);
    _Float16* W2T  = (_Float16*)take((size_t)DFF * DM * 2);
    _Float16* Qh   = (_Float16*)take(MS * DM * 2);       // [b,h,s,hd]
    _Float16* Kh   = (_Float16*)take(MS * DM * 2);       // [b,h,s,hd]
    _Float16* VtH  = (_Float16*)take(MS * DM * 2);       // [b,h,hd,s]
    _Float16* ctxH = (_Float16*)take(MS * DM * 2);       // [b,s,DM]
    float*    aoF  = (float*)   take(MS * DM * 4);       // attn_out
    float*    xF   = (float*)   take(MS * DM * 4);       // after LN1
    _Float16* xH   = (_Float16*)take(MS * DM * 2);
    _Float16* ff1H = (_Float16*)take(MS * DFF * 2);      // ReLU(x@W1+c1)
    float*    ff2F = (float*)   take(MS * DM * 4);       // ff output

    // 1) weight conversion (f32 -> f16, transposed)
    convert_t_kernel<<<(DM * DM + 255) / 256, 256, 0, stream>>>(Wq, WqT, DM, DM);
    convert_t_kernel<<<(DM * DM + 255) / 256, 256, 0, stream>>>(Wk, WkT, DM, DM);
    convert_t_kernel<<<(DM * DM + 255) / 256, 256, 0, stream>>>(Wv, WvT, DM, DM);
    convert_t_kernel<<<(DM * DM + 255) / 256, 256, 0, stream>>>(Wo, WoT, DM, DM);
    convert_t_kernel<<<(DM * DFF + 255) / 256, 256, 0, stream>>>(W1, W1T, DM, DFF);
    convert_t_kernel<<<(DFF * DM + 255) / 256, 256, 0, stream>>>(W2, W2T, DFF, DM);

    // 2) embedding
    embed_kernel<<<(int)(MS * DM / 256), 256, 0, stream>>>(src, We, be, embF, embH);

    // 3) Q/K/V projections (tiles = 512*16 = 8192 -> 1024 blocks of 8 waves)
    gemm16_kernel<<<1024, 256, 0, stream>>>(embH, WqT, bq, Qh,  nullptr, (int)MS, DM, DM, 0);
    gemm16_kernel<<<1024, 256, 0, stream>>>(embH, WkT, bk, Kh,  nullptr, (int)MS, DM, DM, 0);
    gemm16_kernel<<<1024, 256, 0, stream>>>(embH, WvT, bv, VtH, nullptr, (int)MS, DM, DM, 1);

    // 4) energy -> attn region of d_out (pre-softmax)
    energy_kernel<<<dim3(Sc / 16, Sc / 16 / 8, Bc * Hh), 256, 0, stream>>>(Qh, Kh, attn);

    // 5) softmax in place (output tensor #2)
    softmax_kernel<<<Bc * Hh * Sc, 256, 0, stream>>>(attn);

    // 6) ctx = attn @ V, merged-head f16
    ctx_kernel<<<dim3(32, 1, Bc * Hh), 256, 0, stream>>>(attn, VtH, ctxH);

    // 7) O projection -> f32 attn_out
    gemm16_kernel<<<1024, 256, 0, stream>>>(ctxH, WoT, bo, nullptr, aoF, (int)MS, DM, DM, 4);

    // 8) LN1(emb + attn_out) -> x (f32 + f16)
    ln_kernel<<<(int)MS, 256, 0, stream>>>(embF, aoF, g1, b1, xF, xH);

    // 9) FFN: relu(x@W1+c1) -> ff1 (f16); ff1@W2+c2 -> ff2 (f32)
    gemm16_kernel<<<2048, 256, 0, stream>>>(xH, W1T, c1, ff1H, nullptr, (int)MS, DFF, DM, 3);
    gemm16_kernel<<<1024, 256, 0, stream>>>(ff1H, W2T, c2, nullptr, ff2F, (int)MS, DM, DFF, 4);

    // 10) LN2(x + ff) -> out (output tensor #1)
    ln_kernel<<<(int)MS, 256, 0, stream>>>(xF, ff2F, g2, b2, outF, nullptr);
}
